// MatrixAttention_55379308314808
// MI455X (gfx1250) — matrix operand order repaired and checked by reading, not compiled, not run on hardware
//
#include <hip/hip_runtime.h>

// CDNA5 / gfx1250, wave32. Batched GEMM C[b] = A[b] * B[b]^T.
// Pass 1: split fp32 inputs into bf16 hi/lo planes (once; 33.5 MB, L2-resident).
// Pass 2: block-tiled WMMA GEMM. Per 128x128 block and K-chunk of 32, the
//         Tensor Data Mover stages A_hi/A_lo/B_hi/B_lo tiles into LDS
//         (TDM padding -> 80B row stride, bank-conflict-free ds_load_b128),
//         then 8 waves run 3-term split-precision bf16 WMMAs
//         (a*b ~= ah*bh + ah*bl + al*bh) and NT-store the fp32 output.

typedef __attribute__((ext_vector_type(16))) __bf16         v16bf;
typedef __attribute__((ext_vector_type(8)))  float          v8f;
typedef __attribute__((ext_vector_type(4)))  float          f32x4;
typedef __attribute__((ext_vector_type(8)))  unsigned short u16x8;
typedef __attribute__((ext_vector_type(4)))  unsigned short u16x4;
typedef __attribute__((ext_vector_type(4)))  unsigned int   u32x4v;
typedef __attribute__((ext_vector_type(8)))  int            i32x8v;
typedef __attribute__((ext_vector_type(4)))  int            i32x4v;

#define BATCH 8
#define R1M   4096
#define R2N   4096
#define DK    128
#define NELEM ((size_t)BATCH * R1M * DK)   // 4,194,304 elements per matrix

// LDS tile geometry: 128 rows x 32 bf16 (64B) + 16B TDM pad -> 80B row stride.
#define LROW   80
#define LPLANE (128 * LROW)                // 10240 B per plane
// planes: [A_hi][A_lo][B_hi][B_lo] = 40960 B

static __device__ __forceinline__ unsigned short bf_bits(__bf16 b) {
  union { __bf16 b; unsigned short u; } c;
  c.b = b;
  return c.u;
}

// ---------------------------------------------------------------------------
// Pass 1: fp32 -> bf16 hi plane + bf16 lo (residual) plane, 4 elems/thread.
// ---------------------------------------------------------------------------
__global__ __launch_bounds__(256)
void split_bf16_kernel(const float* __restrict__ src,
                       unsigned short* __restrict__ hi,
                       unsigned short* __restrict__ lo) {
  const size_t i4 = ((size_t)blockIdx.x * 256 + threadIdx.x) * 4;
  f32x4 v = __builtin_nontemporal_load((const f32x4*)(src + i4));  // read-once
  u16x4 h, l;
#pragma unroll
  for (int t = 0; t < 4; ++t) {
    __bf16 hb = (__bf16)v[t];           // RNE
    float  r  = v[t] - (float)hb;       // exact residual
    h[t] = bf_bits(hb);
    l[t] = bf_bits((__bf16)r);
  }
  *(u16x4*)(hi + i4) = h;               // RT: keep planes in L2
  *(u16x4*)(lo + i4) = l;
}

// ---------------------------------------------------------------------------
// TDM: 2-D tile load, 128 rows x 32 bf16, row stride DK elems, +16B LDS pad/row.
// D# packing per CDNA5 ISA 8.3/8.4 (group0 128b, group1 256b; groups 2/3 zero).
// This toolchain's builtin is the 6-arg form:
//   (u32x4 g0, i32x8 g1, i32x4 g2, i32x4 g3, i32x8 extra, i32 cpol)
// ---------------------------------------------------------------------------
static __device__ __forceinline__ void tdm_load_tile(unsigned int lds_off,
                                                     const void* gptr) {
  const unsigned long long ga = (unsigned long long)(size_t)gptr;
  u32x4v g0;
  g0[0] = 1u;                                   // count=1, no gather
  g0[1] = lds_off;                              // lds_addr (bytes)
  g0[2] = (unsigned int)ga;                     // global_addr[31:0]
  g0[3] = (unsigned int)((ga >> 32) & 0x01FFFFFFu) | (2u << 30); // [56:32]|type=2

  i32x8v g1;
  g1[0] = (int)((1u << 16)        // data_size = 1 -> 2 bytes
              | (1u << 20)        // pad_enable
              | (3u << 22)        // pad_interval: code 3 -> every 64 B
              | (3u << 25));      // pad_amount:  code 3 -> 4 DWORDs = 16 B
  g1[1] = (int)(32u << 16);       // tensor_dim0 = 32  (bits 79:48 -> w1[31:16])
  g1[2] = (int)(128u << 16);      // tensor_dim1 = 128 (bits 111:80 -> w2[31:16])
  g1[3] = (int)(32u << 16);       // tile_dim0  = 32   (bits 127:112)
  g1[4] = 128;                    // tile_dim1  = 128  (bits 143:128)
  g1[5] = 128;                    // tensor_dim0_stride = 128 elems (bits 207:160)
  g1[6] = 0;
  g1[7] = 0;

  i32x4v z4 = {0, 0, 0, 0};       // groups 2/3 unused (2-D tensor)
  i32x8v z8 = {0, 0, 0, 0, 0, 0, 0, 0};
  __builtin_amdgcn_tensor_load_to_lds(g0, g1, z4, z4, z8, 0);
}

// ---------------------------------------------------------------------------
// Pass 2 helpers
// ---------------------------------------------------------------------------
union FragU {
  u16x8 p[2];
  v16bf v;
};

static __device__ __forceinline__ v8f wmma3(const v16bf& ah, const v16bf& al,
                                            const v16bf& bh, const v16bf& bl,
                                            v8f acc) {
  acc = __builtin_amdgcn_wmma_f32_16x16x32_bf16(false, al, false, bh,
                                                (short)0, acc, false, false);
  acc = __builtin_amdgcn_wmma_f32_16x16x32_bf16(false, ah, false, bl,
                                                (short)0, acc, false, false);
  acc = __builtin_amdgcn_wmma_f32_16x16x32_bf16(false, ah, false, bh,
                                                (short)0, acc, false, false);
  return acc;
}

// ---------------------------------------------------------------------------
// Pass 2: GEMM. Block 256 thr = 8 waves; block tile 128(M) x 128(N).
// Waves 4(M) x 2(N); each wave a 32x64 tile = 2x4 WMMA accumulators.
// ---------------------------------------------------------------------------
__global__ __launch_bounds__(256)
void bmm_wmma_tdm_kernel(const unsigned short* __restrict__ Ahi,
                         const unsigned short* __restrict__ Alo,
                         const unsigned short* __restrict__ Bhi,
                         const unsigned short* __restrict__ Blo,
                         float* __restrict__ C) {
  __shared__ __align__(16) char smem[4 * LPLANE];   // 40 KB

  const int lane = threadIdx.x & 31;
  const int wave = threadIdx.x >> 5;
  const int wm   = wave & 3;      // 0..3 (M groups)
  const int wN   = wave >> 2;     // 0..1 (N groups)
  const int b    = blockIdx.z;

  const int rowB = blockIdx.y * 128;             // block M base
  const int colB = blockIdx.x * 128;             // block N base
  const int row0 = rowB + wm * 32;               // wave M base (32 rows)
  const int col0 = colB + wN * 64;               // wave N base (64 cols)

  const size_t abase = (size_t)b * R1M * DK + (size_t)rowB * DK;
  const size_t bbase = (size_t)b * R2N * DK + (size_t)colB * DK;

  const int l15  = lane & 15;
  const int half = lane >> 4;

  const unsigned int smem_base = (unsigned int)(size_t)&smem[0];
  const char* Ah_s = smem + 0 * LPLANE;
  const char* Al_s = smem + 1 * LPLANE;
  const char* Bh_s = smem + 2 * LPLANE;
  const char* Bl_s = smem + 3 * LPLANE;

  v8f acc[2][4];
#pragma unroll
  for (int i = 0; i < 2; ++i)
#pragma unroll
    for (int j = 0; j < 4; ++j)
      acc[i][j] = (v8f){};

  for (int kc = 0; kc < DK; kc += 32) {
    __syncthreads();   // previous chunk fully consumed before TDM overwrites
    if (wave == 0) {
      tdm_load_tile(smem_base + 0 * LPLANE, Ahi + abase + kc);
      tdm_load_tile(smem_base + 1 * LPLANE, Alo + abase + kc);
      tdm_load_tile(smem_base + 2 * LPLANE, Bhi + bbase + kc);
      tdm_load_tile(smem_base + 3 * LPLANE, Blo + bbase + kc);
      __builtin_amdgcn_s_wait_tensorcnt(0);
    }
    __syncthreads();   // tiles visible to all waves

    // A fragments (16x32 bf16 ISA layout): elems 0..7 = K kb..kb+7,
    // elems 8..15 = K kb+16..kb+23, kb = half*8 within the chunk.
    v16bf ah[2], al[2];
#pragma unroll
    for (int i = 0; i < 2; ++i) {
      const int r = wm * 32 + i * 16 + l15;
      const int o = r * LROW + half * 16;
      FragU h, l;
      h.p[0] = *(const u16x8*)(Ah_s + o);
      h.p[1] = *(const u16x8*)(Ah_s + o + 32);
      l.p[0] = *(const u16x8*)(Al_s + o);
      l.p[1] = *(const u16x8*)(Al_s + o + 32);
      ah[i] = h.v;
      al[i] = l.v;
    }
    // B fragments (32x16 bf16 ISA layout): elems 0..7 = K kb..kb+7, elems 8..15 = K kb+16..kb+23, kb = half*8.
    v16bf bh[4], bl[4];
#pragma unroll
    for (int j = 0; j < 4; ++j) {
      const int r = wN * 64 + j * 16 + l15;
      const int o = r * LROW + half * 16;
      FragU h, l;
      h.p[0] = *(const u16x8*)(Bh_s + o);
      h.p[1] = *(const u16x8*)(Bh_s + o + 32);
      l.p[0] = *(const u16x8*)(Bl_s + o);
      l.p[1] = *(const u16x8*)(Bl_s + o + 32);
      bh[j] = h.v;
      bl[j] = l.v;
    }
#pragma unroll
    for (int i = 0; i < 2; ++i)
#pragma unroll
      for (int j = 0; j < 4; ++j)
        acc[i][j] = wmma3(ah[i], al[i], bh[j], bl[j], acc[i][j]);
  }

  // Store: VGPR r -> lanes 0-15: M=r, N=lane; lanes 16-31: M=r+8, N=lane-16.
  // NT stores: output streams once; keep the bf16 planes resident in L2.
#pragma unroll
  for (int i = 0; i < 2; ++i) {
#pragma unroll
    for (int j = 0; j < 4; ++j) {
      const int n     = col0 + j * 16 + l15;
      const int mbase = row0 + i * 16 + half * 8;
      float* cp = C + ((size_t)b * R1M + mbase) * R2N + n;
#pragma unroll
      for (int r = 0; r < 8; ++r) {
        __builtin_nontemporal_store(acc[i][j][r], cp + (size_t)r * R2N);
      }
    }
  }
}

// ---------------------------------------------------------------------------
// Fallback (insufficient workspace): in-kernel conversion, 2x2 wave tile.
// ---------------------------------------------------------------------------
static __device__ __forceinline__ void cvt16_hilo(const float* __restrict__ f,
                                                  v16bf& hi, v16bf& lo) {
#pragma unroll
  for (int i = 0; i < 16; ++i) {
    __bf16 h = (__bf16)f[i];
    hi[i] = h;
    lo[i] = (__bf16)(f[i] - (float)h);
  }
}

__global__ __launch_bounds__(256)
void bmm_wmma_inline_kernel(const float* __restrict__ A,
                            const float* __restrict__ B,
                            float* __restrict__ C) {
  const int lane = threadIdx.x & 31;
  const int wave = threadIdx.x >> 5;
  const int wm   = wave & 3;
  const int wN   = wave >> 2;
  const int b    = blockIdx.z;

  const int row0 = blockIdx.y * 128 + wm * 32;
  const int col0 = blockIdx.x * 64  + wN * 32;

  const float* __restrict__ Ab = A + (size_t)b * R1M * DK;
  const float* __restrict__ Bb = B + (size_t)b * R2N * DK;

  const int l15  = lane & 15;
  const int half = lane >> 4;

  v8f acc[2][2];
#pragma unroll
  for (int i = 0; i < 2; ++i)
#pragma unroll
    for (int j = 0; j < 2; ++j)
      acc[i][j] = (v8f){};

#pragma unroll
  for (int kc = 0; kc < DK; kc += 32) {
    v16bf ah[2], al[2];
#pragma unroll
    for (int i = 0; i < 2; ++i) {
      const float* p = Ab + (size_t)(row0 + i * 16 + l15) * DK + kc + half * 8;
      float f[16];
      f32x4 v0 = *(const f32x4*)(p + 0);
      f32x4 v1 = *(const f32x4*)(p + 4);
      f32x4 v2 = *(const f32x4*)(p + 16);
      f32x4 v3 = *(const f32x4*)(p + 20);
#pragma unroll
      for (int t = 0; t < 4; ++t) {
        f[t] = v0[t]; f[4 + t] = v1[t]; f[8 + t] = v2[t]; f[12 + t] = v3[t];
      }
      cvt16_hilo(f, ah[i], al[i]);
    }
    v16bf bh[2], bl[2];
#pragma unroll
    for (int j = 0; j < 2; ++j) {
      const float* p = Bb + (size_t)(col0 + j * 16 + l15) * DK + kc + half * 8;
      float f[16];
      f32x4 v0 = *(const f32x4*)(p + 0);
      f32x4 v1 = *(const f32x4*)(p + 4);
      f32x4 v2 = *(const f32x4*)(p + 16);
      f32x4 v3 = *(const f32x4*)(p + 20);
#pragma unroll
      for (int t = 0; t < 4; ++t) {
        f[t] = v0[t]; f[4 + t] = v1[t]; f[8 + t] = v2[t]; f[12 + t] = v3[t];
      }
      cvt16_hilo(f, bh[j], bl[j]);
    }
#pragma unroll
    for (int i = 0; i < 2; ++i)
#pragma unroll
      for (int j = 0; j < 2; ++j)
        acc[i][j] = wmma3(ah[i], al[i], bh[j], bl[j], acc[i][j]);
  }

#pragma unroll
  for (int i = 0; i < 2; ++i) {
#pragma unroll
    for (int j = 0; j < 2; ++j) {
      const int n     = col0 + j * 16 + l15;
      const int mbase = row0 + i * 16 + half * 8;
      float* cp = C + ((size_t)b * R1M + mbase) * R2N + n;
#pragma unroll
      for (int r = 0; r < 8; ++r) {
        __builtin_nontemporal_store(acc[i][j][r], cp + (size_t)r * R2N);
      }
    }
  }
}

// ---------------------------------------------------------------------------
extern "C" void kernel_launch(void* const* d_in, const int* in_sizes, int n_in,
                              void* d_out, int out_size, void* d_ws, size_t ws_size,
                              hipStream_t stream) {
  (void)in_sizes; (void)n_in; (void)out_size;
  const float* A = (const float*)d_in[0];  // (8, 4096, 128) fp32
  const float* B = (const float*)d_in[1];  // (8, 4096, 128) fp32
  float* C = (float*)d_out;                // (8, 4096, 4096) fp32

  const size_t plane = NELEM * sizeof(unsigned short);   // 8,388,608 B
  const size_t need  = 4 * plane;                        // 33,554,432 B

  if (ws_size >= need) {
    unsigned short* Ahi = (unsigned short*)d_ws;
    unsigned short* Alo = Ahi + NELEM;
    unsigned short* Bhi = Alo + NELEM;
    unsigned short* Blo = Bhi + NELEM;

    const int cblocks = (int)(NELEM / (256 * 4));        // 4096
    split_bf16_kernel<<<cblocks, 256, 0, stream>>>(A, Ahi, Alo);
    split_bf16_kernel<<<cblocks, 256, 0, stream>>>(B, Bhi, Blo);

    dim3 grid(R2N / 128, R1M / 128, BATCH);              // (32, 32, 8)
    bmm_wmma_tdm_kernel<<<grid, dim3(256), 0, stream>>>(Ahi, Alo, Bhi, Blo, C);
  } else {
    dim3 grid(R2N / 64, R1M / 128, BATCH);               // (64, 32, 8)
    bmm_wmma_inline_kernel<<<grid, dim3(256), 0, stream>>>(A, B, C);
  }
}
